// Transducer_ASR_27367531610622
// MI455X (gfx1250) — matrix operand order repaired and checked by reading, not compiled, not run on hardware
//
#include <hip/hip_runtime.h>
#include <hip/hip_bf16.h>

// Problem constants from the reference
constexpr int T_  = 200;
constexpr int U_  = 100;
constexpr int U1_ = 101;
constexpr int EE_ = 640;   // encoder/decoder feature dim (both 640)
constexpr int NH_ = 512;   // joint hidden
constexpr int V_  = 128;   // vocab
constexpr int UT_ = (U1_ + 15) / 16;  // 7 u-tiles of 16
constexpr float NEGF = -1000000000.0f;

typedef __attribute__((ext_vector_type(16))) __bf16 v16bf;
typedef __attribute__((ext_vector_type(8)))  __bf16 v8bf;
typedef __attribute__((ext_vector_type(8)))  float  v8f;
typedef __attribute__((ext_vector_type(4)))  float  v4f;

union V16U { v16bf v; v8bf h[2]; };

// Load 8 consecutive f32 and convert to bf16x8 (two aligned 16B loads)
__device__ inline v8bf cvt8(const float* __restrict__ p) {
    v4f a = *(const v4f*)p;
    v4f b = *(const v4f*)(p + 4);
    v8bf r = { (__bf16)a[0], (__bf16)a[1], (__bf16)a[2], (__bf16)a[3],
               (__bf16)b[0], (__bf16)b[1], (__bf16)b[2], (__bf16)b[3] };
    return r;
}

// ---------------------------------------------------------------------------
// Pack: Bt[n*K + k] = (bf16) W[k*Ncols + n]   (transpose + downconvert)
// ---------------------------------------------------------------------------
__global__ void k_transpose_bf16(const float* __restrict__ W, __bf16* __restrict__ Bt,
                                 int K, int Ncols) {
    int idx = blockIdx.x * blockDim.x + threadIdx.x;
    if (idx >= K * Ncols) return;
    int n = idx / K;
    int k = idx - n * K;
    Bt[idx] = (__bf16)W[(size_t)k * Ncols + n];
}

// ---------------------------------------------------------------------------
// C[M,N] = A[M,K] (f32 row-major, cvt->bf16) x B (via Bt[N,K] bf16 row-major)
// fp32 accumulate via v_wmma_f32_16x16x32_bf16. Block = 128 threads = 4 waves,
// each wave owns one 16x16 output tile (grid.y covers N/64).
// ---------------------------------------------------------------------------
__global__ void __launch_bounds__(128)
k_gemm_bf16(const float* __restrict__ A, const __bf16* __restrict__ Bt,
            float* __restrict__ C, int M, int Ncols, int K) {
    const int lane = threadIdx.x & 31;
    const int wave = threadIdx.x >> 5;
    const int r  = lane & 15;   // row (A) / col (B/D)
    const int hi = lane >> 4;   // K-half selector
    const int m0 = blockIdx.x * 16;
    const int n0 = blockIdx.y * 64 + wave * 16;

    int mrow = m0 + r; if (mrow >= M) mrow = M - 1;        // clamp (guarded store)
    const float*  arow = A  + (size_t)mrow * K;
    const __bf16* brow = Bt + (size_t)(n0 + r) * K;

    v8f acc = {0.f, 0.f, 0.f, 0.f, 0.f, 0.f, 0.f, 0.f};
    for (int kt = 0; kt < K / 32; ++kt) {
        const int ka = kt * 32 + hi * 8;     // A lane K-chunks: {ka..ka+7, ka+16..ka+23}
        const int kb = kt * 32 + hi * 8;     // B lane K-chunks: {kb..kb+7, kb+16..kb+23}
        V16U a, b;
        a.h[0] = cvt8(arow + ka);
        a.h[1] = cvt8(arow + ka + 16);
        b.h[0] = *(const v8bf*)(brow + kb);
        b.h[1] = *(const v8bf*)(brow + kb + 16);
        acc = __builtin_amdgcn_wmma_f32_16x16x32_bf16(false, a.v, false, b.v,
                                                      (short)0, acc, false, false);
    }
#pragma unroll
    for (int g = 0; g < 8; ++g) {
        int row = m0 + g + 8 * hi;           // D layout: VGPR g -> M = g + 8*hi
        if (row < M) C[(size_t)row * Ncols + n0 + r] = acc[g];
    }
}

// ---------------------------------------------------------------------------
// Fused joint: one block per (n, t, u-tile of 16).
//   h[i,k] = tanh(enc_p[n,t,k] + dec_p[n,u0+i,k] + b_joint[k])  -> LDS (bf16)
//   logits = h @ W_out + b_out (WMMA, 4 waves x 2 N-tiles x 16 K-steps)
//   log-softmax over V=128; emit blank_lp and emit_lp only.
// ---------------------------------------------------------------------------
__global__ void __launch_bounds__(128)
k_joint(const float* __restrict__ enc_p, const float* __restrict__ dec_p,
        const float* __restrict__ b_joint, const __bf16* __restrict__ Wot,
        const float* __restrict__ b_out, const int* __restrict__ targets,
        float* __restrict__ blank_out, float* __restrict__ emit_out) {
    constexpr int AS = NH_ + 8;              // 520: 4-bank stagger per row
    __shared__ __bf16 hA[16 * AS];
    __shared__ float  lg[16 * V_];

    const int b  = blockIdx.x;
    const int ut = b % UT_;
    const int nt = b / UT_;                  // n*T + t
    const int n  = nt / T_;
    const int u0 = ut * 16;

    // ---- build 16x512 tanh tile in bf16 ----
    const float* ep = enc_p + (size_t)nt * NH_;
    for (int idx = threadIdx.x; idx < 16 * NH_; idx += 128) {
        const int i = idx >> 9;              // NH_ = 512
        const int k = idx & (NH_ - 1);
        int u = u0 + i; if (u > U1_ - 1) u = U1_ - 1;   // clamp pad rows
        float hv = tanhf(ep[k] + dec_p[((size_t)n * U1_ + u) * NH_ + k] + b_joint[k]);
        hA[i * AS + k] = (__bf16)hv;
    }
    __syncthreads();

    // ---- WMMA: wave w handles output columns [32w, 32w+32) ----
    const int lane = threadIdx.x & 31;
    const int wave = threadIdx.x >> 5;
    const int r  = lane & 15;
    const int hi = lane >> 4;
    v8f acc0 = {0.f,0.f,0.f,0.f,0.f,0.f,0.f,0.f};
    v8f acc1 = {0.f,0.f,0.f,0.f,0.f,0.f,0.f,0.f};
    const __bf16* arow  = hA  + r * AS;
    const __bf16* b0row = Wot + (size_t)(wave * 32 + r) * NH_;
    const __bf16* b1row = b0row + (size_t)16 * NH_;
#pragma unroll
    for (int kt = 0; kt < NH_ / 32; ++kt) {
        const int ka = kt * 32 + hi * 8;
        const int kb = kt * 32 + hi * 8;
        V16U a, bb0, bb1;
        a.h[0]   = *(const v8bf*)(arow + ka);
        a.h[1]   = *(const v8bf*)(arow + ka + 16);
        bb0.h[0] = *(const v8bf*)(b0row + kb);
        bb0.h[1] = *(const v8bf*)(b0row + kb + 16);
        bb1.h[0] = *(const v8bf*)(b1row + kb);
        bb1.h[1] = *(const v8bf*)(b1row + kb + 16);
        acc0 = __builtin_amdgcn_wmma_f32_16x16x32_bf16(false, a.v, false, bb0.v,
                                                       (short)0, acc0, false, false);
        acc1 = __builtin_amdgcn_wmma_f32_16x16x32_bf16(false, a.v, false, bb1.v,
                                                       (short)0, acc1, false, false);
    }
#pragma unroll
    for (int g = 0; g < 8; ++g) {
        const int row = g + 8 * hi;
        const int c0  = wave * 32 + r;
        lg[row * V_ + c0]      = acc0[g] + b_out[c0];
        lg[row * V_ + c0 + 16] = acc1[g] + b_out[c0 + 16];
    }
    __syncthreads();

    // ---- log-softmax readout: 8 lanes per row ----
    const int row = threadIdx.x >> 3;
    const int sub = threadIdx.x & 7;
    const float* lrow = lg + row * V_;
    float mx = -3.4e38f;
    for (int c = sub; c < V_; c += 8) mx = fmaxf(mx, lrow[c]);
    for (int off = 4; off; off >>= 1) mx = fmaxf(mx, __shfl_xor(mx, off, 8));
    float s = 0.f;
    for (int c = sub; c < V_; c += 8) s += __expf(lrow[c] - mx);
    for (int off = 4; off; off >>= 1) s += __shfl_xor(s, off, 8);
    if (sub == 0) {
        const int u = u0 + row;
        if (u < U1_) {
            const float lse = mx + __logf(s);
            blank_out[(size_t)nt * U1_ + u] = lrow[0] - lse;       // lp[..., BLANK]
            if (u < U_) {
                const int tg = targets[n * U_ + u];
                emit_out[(size_t)nt * U_ + u] = lrow[tg] - lse;    // lp[..., targets]
            }
        }
    }
}

// ---------------------------------------------------------------------------
// RNN-T DP, anti-diagonal wavefront. Thread u owns column u; C[t,u] =
// logaddexp(C[t-1,u]+blank[t-1,u], C[t,u-1]+emit[t,u-1]); ll at (lenT-1, lenU).
// ---------------------------------------------------------------------------
__global__ void __launch_bounds__(128)
k_dp(const float* __restrict__ blank, const float* __restrict__ emit,
     const int* __restrict__ inputs_len, const int* __restrict__ targets_len,
     float* __restrict__ llbuf) {
    __shared__ float left[128];
    const int n = blockIdx.x;
    const int u = threadIdx.x;
    const float* Bn = blank + (size_t)n * T_ * U1_;
    const float* En = emit  + (size_t)n * T_ * U_;
    const int lenT = inputs_len[n];
    const int lenU = targets_len[n];

    float Cown = NEGF;
    left[u] = NEGF;
    __syncthreads();

    for (int d = 0; d < T_ + U1_ - 1; ++d) {
        const int t = d - u;
        const bool active = (u < U1_) && (t >= 0) && (t < T_);
        const float lprev = (u > 0) ? left[u - 1] : NEGF;  // C[t, u-1] from diag d-1
        float Cnew = NEGF;
        if (active) {
            const float Bv = (t == 0) ? ((u == 0) ? 0.f : NEGF)
                                      : (Cown + Bn[(size_t)(t - 1) * U1_ + u]);
            if (u == 0) {
                Cnew = Bv;
            } else {
                const float Ev = lprev + En[(size_t)t * U_ + (u - 1)];
                const float m  = fmaxf(Bv, Ev);
                Cnew = m + log1pf(__expf(fminf(Bv, Ev) - m));   // logaddexp
            }
            Cown = Cnew;
            if (t == lenT - 1 && u == lenU)
                llbuf[n] = Cnew + Bn[(size_t)t * U1_ + u];
        }
        __syncthreads();
        if (active) left[u] = Cnew;
        __syncthreads();
    }
}

__global__ void k_finalize(const float* __restrict__ llbuf, float* __restrict__ out, int n) {
    if (threadIdx.x == 0) {
        float s = 0.f;
        for (int i = 0; i < n; ++i) s += llbuf[i];
        out[0] = -s / (float)n;
    }
}

// ---------------------------------------------------------------------------
extern "C" void kernel_launch(void* const* d_in, const int* in_sizes, int n_in,
                              void* d_out, int out_size, void* d_ws, size_t ws_size,
                              hipStream_t stream) {
    (void)n_in; (void)out_size; (void)ws_size;
    const float* enc  = (const float*)d_in[0];   // (N,T,640)
    const float* dec  = (const float*)d_in[1];   // (N,U1,640)
    const float* Wj   = (const float*)d_in[2];   // (1280,512)
    const float* bj   = (const float*)d_in[3];   // (512)
    const float* Wo   = (const float*)d_in[4];   // (512,128)
    const float* bo   = (const float*)d_in[5];   // (128)
    const int*   tgt  = (const int*)d_in[6];     // (N,U)
    const int*   ilen = (const int*)d_in[7];     // (N)
    const int*   tlen = (const int*)d_in[8];     // (N)
    const int NB = in_sizes[7];                  // batch size

    // --- carve workspace (256B aligned segments) ---
    char* p = (char*)d_ws;
    auto carve = [&](size_t bytes) {
        char* r = p; p += (bytes + 255) & ~(size_t)255; return r;
    };
    float*  enc_p = (float*)carve((size_t)NB * T_  * NH_ * sizeof(float));
    float*  dec_p = (float*)carve((size_t)NB * U1_ * NH_ * sizeof(float));
    __bf16* Wjt_e = (__bf16*)carve((size_t)NH_ * EE_ * sizeof(__bf16));
    __bf16* Wjt_d = (__bf16*)carve((size_t)NH_ * EE_ * sizeof(__bf16));
    __bf16* Wot   = (__bf16*)carve((size_t)V_  * NH_ * sizeof(__bf16));
    float*  blank = (float*)carve((size_t)NB * T_ * U1_ * sizeof(float));
    float*  emit  = (float*)carve((size_t)NB * T_ * U_  * sizeof(float));
    float*  llbuf = (float*)carve((size_t)NB * sizeof(float));

    // --- pack weights (transpose + bf16) ---
    {
        int n1 = EE_ * NH_;
        k_transpose_bf16<<<(n1 + 255) / 256, 256, 0, stream>>>(Wj, Wjt_e, EE_, NH_);
        k_transpose_bf16<<<(n1 + 255) / 256, 256, 0, stream>>>(Wj + (size_t)EE_ * NH_,
                                                               Wjt_d, EE_, NH_);
        int n2 = NH_ * V_;
        k_transpose_bf16<<<(n2 + 255) / 256, 256, 0, stream>>>(Wo, Wot, NH_, V_);
    }

    // --- projections: enc_p = enc @ Wj_enc ; dec_p = dec @ Wj_dec ---
    {
        dim3 g1((NB * T_ + 15) / 16, NH_ / 64);
        k_gemm_bf16<<<g1, 128, 0, stream>>>(enc, Wjt_e, enc_p, NB * T_, NH_, EE_);
        dim3 g2((NB * U1_ + 15) / 16, NH_ / 64);
        k_gemm_bf16<<<g2, 128, 0, stream>>>(dec, Wjt_d, dec_p, NB * U1_, NH_, EE_);
    }

    // --- fused tanh + joint GEMM + log-softmax -> blank/emit log-probs ---
    k_joint<<<NB * T_ * UT_, 128, 0, stream>>>(enc_p, dec_p, bj, Wot, bo, tgt,
                                               blank, emit);

    // --- RNN-T wavefront DP + mean ---
    k_dp<<<NB, 128, 0, stream>>>(blank, emit, ilen, tlen, llbuf);
    k_finalize<<<1, 32, 0, stream>>>(llbuf, (float*)d_out, NB);
}
